// MatrixModel_24180665876948
// MI455X (gfx1250) — matrix operand order repaired and checked by reading, not compiled, not run on hardware
//
#include <hip/hip_runtime.h>
#include <hip/hip_bf16.h>

// ---------------------------------------------------------------------------
// Model dims (fixed by the reference)
// ---------------------------------------------------------------------------
#define RWS   2048   // B*S tokens
#define HDIM  2048
#define KDDIM 1024
#define IDIM  8192
#define NLAY  2
#define NCOARSE 8192
#define TOPK  64

// GEMM tiling
#define BM 128
#define BN 128
#define BK 32
#define LDT 40      // padded LDS row length in u16 elems (80B = 5x16B, keeps b128 aligned)

typedef __bf16 v16bf __attribute__((ext_vector_type(16)));
typedef float  v8f   __attribute__((ext_vector_type(8)));
typedef unsigned short u16;
typedef u16 u16x8 __attribute__((ext_vector_type(8)));
typedef u16 u16x4 __attribute__((ext_vector_type(4)));

union FragBF {
  v16bf bf;
  u16x8 u[2];
};

__device__ __forceinline__ u16 f2bf(float f) {
  unsigned u = __float_as_uint(f);
  unsigned r = u + 0x7fffu + ((u >> 16) & 1u);   // round-to-nearest-even
  return (u16)(r >> 16);
}

__device__ __forceinline__ void async_b128(unsigned ldsoff, const u16* g) {
  asm volatile("global_load_async_to_lds_b128 %0, %1, off"
               :: "v"(ldsoff), "v"(g) : "memory");
}

__device__ __forceinline__ void wait_async0() {
  asm volatile("s_wait_asynccnt 0x0" ::: "memory");
}

// Fragment gathers + 8 WMMAs for one staged 128x32 K-tile (static buffer base).
__device__ __forceinline__ void compute_tile(const u16* asb, const u16* bsb,
                                             int waveM, int waveN, int laneM, int hi,
                                             v8f (&acc)[2][4]) {
  FragBF a[2], b[4];
  // A fragment (16x32 bf16): lanes 0-15 hold K {0..7,16..23}, lanes 16-31 {8..15,24..31}
#pragma unroll
  for (int i = 0; i < 2; i++) {
    const u16* p = asb + (waveM + i * 16 + laneM) * LDT;
    a[i].u[0] = *(const u16x8*)(p + hi * 8);
    a[i].u[1] = *(const u16x8*)(p + 16 + hi * 8);
  }
  // B fragment (32x16 bf16): lane n, lanes 0-15 hold K {0..7,16..23}, lanes 16-31 {8..15,24..31}
#pragma unroll
  for (int j = 0; j < 4; j++) {
    const u16* p = bsb + (waveN + j * 16 + laneM) * LDT;
    b[j].u[0] = *(const u16x8*)(p + hi * 8);
    b[j].u[1] = *(const u16x8*)(p + 16 + hi * 8);
  }
#pragma unroll
  for (int i = 0; i < 2; i++)
#pragma unroll
    for (int j = 0; j < 4; j++)
      acc[i][j] = __builtin_amdgcn_wmma_f32_16x16x32_bf16(
          false, a[i].bf, false, b[j].bf, (short)0, acc[i][j], false, false);
}

// ---------------------------------------------------------------------------
// C[M,N] = A[M,K] @ W[N,K]^T with bf16 operands, f32 WMMA accumulation.
// flags bit0: store f32 C;  bit1: multiply acc by existing f32 C first.
// Cbf != null: also store bf16 mirror (feeds the next GEMM's A operand).
// M,N multiples of 128; K multiple of 64 (K/BK even; true for 1024/2048/8192).
// ---------------------------------------------------------------------------
__global__ __launch_bounds__(256) void gemm_wmma_bf16(
    const u16* __restrict__ A, const u16* __restrict__ W,
    float* __restrict__ C, u16* __restrict__ Cbf,
    int N, int K, int flags) {
  __shared__ u16 As[2][BM * LDT];
  __shared__ u16 Bs[2][BN * LDT];

  const int m0   = blockIdx.y * BM;
  const int n0   = blockIdx.x * BN;
  const int tid  = threadIdx.x;
  const int lane = tid & 31;
  const int wid  = tid >> 5;
  const int waveM = (wid & 3) * 32;   // 4 waves along M
  const int waveN = (wid >> 2) * 64;  // 2 waves along N
  const int laneM = lane & 15;
  const int hi    = lane >> 4;

  // ---- async staging state: per-thread 16B chunk pointers, bumped by BK/tile
  const int c0 = tid, c1 = tid + 256;           // chunk ids 0..511 of a 128x32 tile
  const int r0 = c0 >> 2, q0 = c0 & 3;
  const int r1 = c1 >> 2, q1 = c1 & 3;
  const u16* gA0 = A + (size_t)(m0 + r0) * (size_t)K + q0 * 8;
  const u16* gA1 = A + (size_t)(m0 + r1) * (size_t)K + q1 * 8;
  const u16* gW0 = W + (size_t)(n0 + r0) * (size_t)K + q0 * 8;
  const u16* gW1 = W + (size_t)(n0 + r1) * (size_t)K + q1 * 8;
  const unsigned dA0 = r0 * LDT + q0 * 8;       // LDS element offsets within a tile
  const unsigned dA1 = r1 * LDT + q1 * 8;
  const unsigned lA0[2] = { (unsigned)(uintptr_t)(&As[0][0] + dA0), (unsigned)(uintptr_t)(&As[1][0] + dA0) };
  const unsigned lA1[2] = { (unsigned)(uintptr_t)(&As[0][0] + dA1), (unsigned)(uintptr_t)(&As[1][0] + dA1) };
  const unsigned lB0[2] = { (unsigned)(uintptr_t)(&Bs[0][0] + dA0), (unsigned)(uintptr_t)(&Bs[1][0] + dA0) };
  const unsigned lB1[2] = { (unsigned)(uintptr_t)(&Bs[0][0] + dA1), (unsigned)(uintptr_t)(&Bs[1][0] + dA1) };

  // stage the next K-tile into buffer `b`, advance chunk pointers
#define STAGE(b)                                                   \
  do {                                                             \
    async_b128(lA0[b], gA0); async_b128(lA1[b], gA1);              \
    async_b128(lB0[b], gW0); async_b128(lB1[b], gW1);              \
    gA0 += BK; gA1 += BK; gW0 += BK; gW1 += BK;                    \
  } while (0)

  v8f acc[2][4];
  const v8f vzero = {0.f, 0.f, 0.f, 0.f, 0.f, 0.f, 0.f, 0.f};
#pragma unroll
  for (int i = 0; i < 2; i++)
#pragma unroll
    for (int j = 0; j < 4; j++) acc[i][j] = vzero;

  const int T = K / BK;          // even for all GEMMs in this model

  STAGE(0);
  wait_async0();
  __syncthreads();

  for (int t = 0; t < T; t += 2) {
    // even tile lives in buf0; prefetch odd tile into buf1 (t+1 < T since T even)
    STAGE(1);
    compute_tile(&As[0][0], &Bs[0][0], waveM, waveN, laneM, hi, acc);
    wait_async0();
    __syncthreads();

    // odd tile lives in buf1; prefetch next even tile into buf0
    if (t + 2 < T) STAGE(0);
    compute_tile(&As[1][0], &Bs[1][0], waveM, waveN, laneM, hi, acc);
    wait_async0();
    __syncthreads();
  }
#undef STAGE

  // Epilogue: C/D layout — lane(0..15)=N col, vgpr r -> M row r (lanes<16) / 8+r
  const bool do_mul = (flags & 2) != 0;
  const bool do_f32 = (flags & 1) != 0;
  const bool do_bf  = (Cbf != nullptr);
#pragma unroll
  for (int i = 0; i < 2; i++) {
#pragma unroll
    for (int j = 0; j < 4; j++) {
#pragma unroll
      for (int r = 0; r < 8; r++) {
        const int gm = m0 + waveM + i * 16 + hi * 8 + r;
        const int gn = n0 + waveN + j * 16 + laneM;
        const size_t o = (size_t)gm * (size_t)N + gn;
        float val = acc[i][j][r];
        if (do_mul) val *= C[o];
        if (do_f32) C[o] = val;
        if (do_bf)  Cbf[o] = f2bf(val);
      }
    }
  }
}

// f32 -> bf16 weight conversion (vectorized x4)
__global__ void cvt_f32_bf16(const float* __restrict__ in, u16* __restrict__ out, int total4) {
  int i = blockIdx.x * blockDim.x + threadIdx.x;
  if (i >= total4) return;
  float4 f = ((const float4*)in)[i];
  u16x4 o;
  o[0] = f2bf(f.x); o[1] = f2bf(f.y); o[2] = f2bf(f.z); o[3] = f2bf(f.w);
  ((u16x4*)out)[i] = o;
}

// h[r, :] = embed[ids[r], :]  (f32 + bf16 mirror)
__global__ void gather_embed(const int* __restrict__ ids,
                             const float* __restrict__ embed,
                             float* __restrict__ h, u16* __restrict__ hbf, int total) {
  int i = blockIdx.x * blockDim.x + threadIdx.x;
  if (i >= total) return;
  int r = i >> 11;     // / HDIM
  int c = i & (HDIM - 1);
  float e = embed[(size_t)ids[r] * HDIM + c];
  h[i] = e;
  hbf[i] = f2bf(e);
}

// Fold the attention group broadcast into Wo and convert to bf16:
// Wof[n, kv*128+d] = bf16(Wo[n, kv*256+d] + Wo[n, kv*256+128+d])
__global__ void fold_wo_bf(const float* __restrict__ Wo, u16* __restrict__ Wof, int total) {
  int i = blockIdx.x * blockDim.x + threadIdx.x;
  if (i >= total) return;
  int n = i >> 10;            // / KDDIM
  int t = i & (KDDIM - 1);
  int kv = t >> 7;
  int d  = t & 127;
  const float* row = Wo + (size_t)n * HDIM + kv * 256 + d;
  Wof[i] = f2bf(row[0] + row[128]);
}

// Iterative top-64 per row over NCOARSE logits; sorted descending,
// lowest-index wins ties (matches lax.top_k ordering).
__global__ __launch_bounds__(256) void topk64(const float* __restrict__ coarse,
                                              int* __restrict__ idx_out) {
  __shared__ float sv[NCOARSE];
  __shared__ float rv[256];
  __shared__ int   ri[256];
  const int row = blockIdx.x, tid = threadIdx.x;
  const float* src = coarse + (size_t)row * NCOARSE;
  for (int c = tid; c < NCOARSE; c += 256) sv[c] = src[c];
  __syncthreads();
  for (int sel = 0; sel < TOPK; sel++) {
    float bv = -3.4e38f; int bi = 0x7fffffff;
    for (int c = tid; c < NCOARSE; c += 256) {
      float v = sv[c];
      if (v > bv) { bv = v; bi = c; }      // ascending scan keeps lowest index
    }
    rv[tid] = bv; ri[tid] = bi;
    __syncthreads();
    for (int off = 128; off > 0; off >>= 1) {
      if (tid < off) {
        float ov = rv[tid + off]; int oi = ri[tid + off];
        if (ov > rv[tid] || (ov == rv[tid] && oi < ri[tid])) { rv[tid] = ov; ri[tid] = oi; }
      }
      __syncthreads();
    }
    if (tid == 0) {
      idx_out[(size_t)row * TOPK + sel] = ri[0];
      sv[ri[0]] = -3.4e38f;
    }
    __syncthreads();
  }
}

// logits[row, k] = dot(h[row,:], Wout[idx[row,k],:])  (exact, f32)
__global__ __launch_bounds__(256) void regather_logits(
    const float* __restrict__ h, const float* __restrict__ Wout,
    const int* __restrict__ idx, float* __restrict__ logits) {
  __shared__ float hrow[HDIM];
  __shared__ float red[256];
  const int row = blockIdx.x, tid = threadIdx.x;
  const float* hp = h + (size_t)row * HDIM;
  for (int c = tid; c < HDIM; c += 256) hrow[c] = hp[c];
  __syncthreads();
  for (int kk = 0; kk < TOPK; kk++) {
    const int r = idx[(size_t)row * TOPK + kk];
    const float* w = Wout + (size_t)r * HDIM;
    float s = 0.f;
    for (int c = tid; c < HDIM; c += 256) s += hrow[c] * w[c];
    red[tid] = s;
    __syncthreads();
    for (int off = 128; off > 0; off >>= 1) {
      if (tid < off) red[tid] += red[tid + off];
      __syncthreads();
    }
    if (tid == 0) logits[(size_t)row * TOPK + kk] = red[0];
    __syncthreads();
  }
}

extern "C" void kernel_launch(void* const* d_in, const int* in_sizes, int n_in,
                              void* d_out, int out_size, void* d_ws, size_t ws_size,
                              hipStream_t stream) {
  const int*   ids   = (const int*)d_in[0];
  const float* embed = (const float*)d_in[1];
  // d_in[2] (Wq) and d_in[3] (Wk) are mathematically dead: the reference einsum
  // contracts the softmax weights against a q-independent v, and softmax rows
  // sum to 1, so attention reduces to a group-broadcast of v.
  const float* Wv    = (const float*)d_in[4];
  const float* Wo    = (const float*)d_in[5];
  const float* Wg    = (const float*)d_in[6];
  const float* Wu    = (const float*)d_in[7];
  const float* Wd    = (const float*)d_in[8];
  const float* Wout  = (const float*)d_in[9];

  // Workspace layout (bytes):
  //  h_f32 : 16MB | h_bf : 8MB | v_bf : 4MB | p_bf : 32MB | gate/coarse f32 : 64MB | wbuf_bf : 32MB
  char* ws = (char*)d_ws;
  float* h_f32 = (float*)(ws);
  u16*   h_bf  = (u16*)  (ws + (16ull << 20));
  u16*   v_bf  = (u16*)  (ws + (24ull << 20));
  u16*   p_bf  = (u16*)  (ws + (28ull << 20));
  float* gate  = (float*)(ws + (60ull << 20));   // also reused for coarse logits
  u16*   wbuf  = (u16*)  (ws + (124ull << 20));  // bf16 weight staging (<= 32MB)

  float* logits = (float*)d_out;
  int*   idxo   = (int*)d_out + (size_t)RWS * TOPK;

  const int CVT = 256;
  gather_embed<<<(RWS * HDIM + 255) / 256, 256, 0, stream>>>(ids, embed, h_f32, h_bf, RWS * HDIM);

  for (int l = 0; l < NLAY; l++) {
    const float* Wvl = Wv + (size_t)l * KDDIM * HDIM;
    const float* Wol = Wo + (size_t)l * HDIM * HDIM;
    const float* Wgl = Wg + (size_t)l * IDIM * HDIM;
    const float* Wul = Wu + (size_t)l * IDIM * HDIM;
    const float* Wdl = Wd + (size_t)l * HDIM * IDIM;

    // v_bf = bf16(h @ Wv^T)                 [2048 x 1024, K=2048]
    cvt_f32_bf16<<<(KDDIM * HDIM / 4 + CVT - 1) / CVT, CVT, 0, stream>>>(Wvl, wbuf, KDDIM * HDIM / 4);
    gemm_wmma_bf16<<<dim3(KDDIM / BN, RWS / BM), 256, 0, stream>>>(h_bf, wbuf, gate, v_bf, KDDIM, HDIM, 0);

    // h = v @ Wof^T  (group broadcast folded) [2048 x 2048, K=1024]
    fold_wo_bf<<<(HDIM * KDDIM + CVT - 1) / CVT, CVT, 0, stream>>>(Wol, wbuf, HDIM * KDDIM);
    gemm_wmma_bf16<<<dim3(HDIM / BN, RWS / BM), 256, 0, stream>>>(v_bf, wbuf, h_f32, h_bf, HDIM, KDDIM, 1);

    // gate = h @ Wg^T                        [2048 x 8192, K=2048]
    cvt_f32_bf16<<<(IDIM * HDIM / 4 + CVT - 1) / CVT, CVT, 0, stream>>>(Wgl, wbuf, IDIM * HDIM / 4);
    gemm_wmma_bf16<<<dim3(IDIM / BN, RWS / BM), 256, 0, stream>>>(h_bf, wbuf, gate, (u16*)nullptr, IDIM, HDIM, 1);

    // p_bf = bf16(gate .* (h @ Wu^T))  (fused Hadamard)
    cvt_f32_bf16<<<(IDIM * HDIM / 4 + CVT - 1) / CVT, CVT, 0, stream>>>(Wul, wbuf, IDIM * HDIM / 4);
    gemm_wmma_bf16<<<dim3(IDIM / BN, RWS / BM), 256, 0, stream>>>(h_bf, wbuf, gate, p_bf, IDIM, HDIM, 2);

    // h = p @ Wd^T                           [2048 x 2048, K=8192]
    cvt_f32_bf16<<<(HDIM * IDIM / 4 + CVT - 1) / CVT, CVT, 0, stream>>>(Wdl, wbuf, HDIM * IDIM / 4);
    gemm_wmma_bf16<<<dim3(HDIM / BN, RWS / BM), 256, 0, stream>>>(p_bf, wbuf, h_f32, h_bf, HDIM, IDIM, 1);
  }

  // coarse logits over first 8192 vocab rows (reuse gate buffer)
  cvt_f32_bf16<<<(NCOARSE * HDIM / 4 + CVT - 1) / CVT, CVT, 0, stream>>>(Wout, wbuf, NCOARSE * HDIM / 4);
  gemm_wmma_bf16<<<dim3(NCOARSE / BN, RWS / BM), 256, 0, stream>>>(h_bf, wbuf, gate, (u16*)nullptr, NCOARSE, HDIM, 1);
  // top-64 indices
  topk64<<<RWS, 256, 0, stream>>>(gate, idxo);
  // exact re-gather logits in f32
  regather_logits<<<RWS, 256, 0, stream>>>(h_f32, Wout, idxo, logits);
}
